// GraphAttentionLayer_87557203296588
// MI455X (gfx1250) — matrix operand order repaired and checked by reading, not compiled, not run on hardware
//
#include <hip/hip_runtime.h>
#include <hip/hip_bf16.h>

typedef __bf16 v4bf  __attribute__((ext_vector_type(4)));
typedef __bf16 v8bf  __attribute__((ext_vector_type(8)));
typedef __bf16 v16bf __attribute__((ext_vector_type(16)));
typedef float  v8f   __attribute__((ext_vector_type(8)));

#define GAT_ALPHA 0.2f
#define GAT_NEG_INF (-9.0e15f)

// ---------------------------------------------------------------------------
// prep: c_x = csk_W^T @ a2_SS, c_o = csk_W^T @ a2_DS, k = csk_b . a2_*
// ---------------------------------------------------------------------------
__global__ __launch_bounds__(256) void prep_kernel(
    const float* __restrict__ cskW, const float* __restrict__ cskB,
    const float* __restrict__ aSS,  const float* __restrict__ aDS,
    float* __restrict__ cx, float* __restrict__ co, float* __restrict__ kk,
    int F)
{
  int k = blockIdx.x * 256 + threadIdx.x;
  if (k < F) {
    float sx = 0.f, so = 0.f;
    for (int n = 0; n < F; ++n) {
      float w = cskW[(size_t)n * F + k];           // column k, coalesced
      sx += w * aSS[F + n];
      so += w * aDS[F + n];
    }
    cx[k] = sx;
    co[k] = so;
  }
  if (blockIdx.x == 0 && threadIdx.x == 0) {
    float s0 = 0.f, s1 = 0.f;
    for (int n = 0; n < F; ++n) { s0 += cskB[n] * aSS[F + n]; s1 += cskB[n] * aDS[F + n]; }
    kk[0] = s0; kk[1] = s1;
  }
}

// ---------------------------------------------------------------------------
// Batched 1024x1024x1024 GEMM via v_wmma_f32_16x16x32_bf16, f32 accumulate.
// Block tile 128x128, K-step 32 (one WMMA K). 8 waves (wave32) in a 4x2 grid,
// each owning a 32x64 sub-tile = 2x4 accumulators. Double-buffered LDS:
// next tile's global loads are issued before consuming current fragments,
// one barrier per K-step. Compile-time dims fold C-store addressing into
// immediate offsets.
// ---------------------------------------------------------------------------
template <bool A_BF16, bool B_BF16, bool C_BF16, bool ELU>
__global__ __launch_bounds__(256) void gemm_wmma(
    const void* __restrict__ Aop, const void* __restrict__ Bop,
    void* __restrict__ Cop,
    long long strideA, long long strideB, long long strideC)
{
  constexpr int NN = 1024;            // N == K == 1024 for every GEMM here
  constexpr int KK = 1024;
  constexpr int TSTEPS = KK / 32;

  __shared__ alignas(16) __bf16 As[2][128][40];   // [m][k]  (pad 32->40)
  __shared__ alignas(16) __bf16 Bs[2][128][40];   // [n][k]  transposed B tile

  const int z      = blockIdx.z;
  const int tile_m = blockIdx.y * 128;
  const int tile_n = blockIdx.x * 128;

  const float*  Af = ((const float*) Aop) + (size_t)z * strideA;
  const __bf16* Ab = ((const __bf16*)Aop) + (size_t)z * strideA;
  const float*  Bf = ((const float*) Bop) + (size_t)z * strideB;
  const __bf16* Bb = ((const __bf16*)Bop) + (size_t)z * strideB;
  float*  Cf = ((float*) Cop) + (size_t)z * strideC;
  __bf16* Cb = ((__bf16*)Cop) + (size_t)z * strideC;

  const int tid  = threadIdx.x;
  const int lane = tid & 31;
  const int wave = tid >> 5;
  const int wm0  = (wave >> 1) * 32;   // wave grid 4 x 2
  const int wn0  = (wave & 1) * 64;
  const int lm   = lane & 15;
  const int lh   = lane >> 4;

  // staging assignments
  const int ar = tid >> 3;             // A row 0..31 (+32 steps)
  const int ac = (tid & 7) * 4;        // A col 0..28
  const int bk = tid >> 5;             // B k   0..7  (+8 steps)
  const int bn = (tid & 31) * 4;       // B n   0..124

  float4 fa[4]; v4bf ba[4];
  float4 fb[4]; v4bf bb[4];

  auto globalLoad = [&](int k0) {
#pragma unroll
    for (int q = 0; q < 4; ++q) {
      const int r  = ar + q * 32;
      const int kk = bk + q * 8;
      if constexpr (A_BF16)
        ba[q] = *(const v4bf*)(Ab + (size_t)(tile_m + r) * KK + k0 + ac);
      else
        fa[q] = *(const float4*)(Af + (size_t)(tile_m + r) * KK + k0 + ac);
      if constexpr (B_BF16)
        bb[q] = *(const v4bf*)(Bb + (size_t)(k0 + kk) * NN + tile_n + bn);
      else
        fb[q] = *(const float4*)(Bf + (size_t)(k0 + kk) * NN + tile_n + bn);
    }
  };

  auto ldsStore = [&](int buf) {
#pragma unroll
    for (int q = 0; q < 4; ++q) {
      const int r  = ar + q * 32;
      const int kk = bk + q * 8;
      v4bf av;
      if constexpr (A_BF16) av = ba[q];
      else {
        av[0] = (__bf16)fa[q].x; av[1] = (__bf16)fa[q].y;
        av[2] = (__bf16)fa[q].z; av[3] = (__bf16)fa[q].w;
      }
      *(v4bf*)&As[buf][r][ac] = av;
      __bf16 b0, b1, b2, b3;
      if constexpr (B_BF16) { b0 = bb[q][0]; b1 = bb[q][1]; b2 = bb[q][2]; b3 = bb[q][3]; }
      else { b0 = (__bf16)fb[q].x; b1 = (__bf16)fb[q].y; b2 = (__bf16)fb[q].z; b3 = (__bf16)fb[q].w; }
      Bs[buf][bn + 0][kk] = b0;
      Bs[buf][bn + 1][kk] = b1;
      Bs[buf][bn + 2][kk] = b2;
      Bs[buf][bn + 3][kk] = b3;
    }
  };

  v8f acc[2][4];
#pragma unroll
  for (int i = 0; i < 2; ++i)
#pragma unroll
    for (int j = 0; j < 4; ++j) acc[i][j] = (v8f)0.0f;

  globalLoad(0);
  ldsStore(0);
  __syncthreads();

  for (int t = 0; t < TSTEPS; ++t) {
    const int cur = t & 1;
    if (t + 1 < TSTEPS) globalLoad((t + 1) * 32);   // prefetch next tile

    // A 16x32 frag: lane half h -> K {8h..8h+7} in elems 0-7, {16+8h..} in 8-15
    v16bf afrag[2];
#pragma unroll
    for (int i = 0; i < 2; ++i) {
      const __bf16* p = &As[cur][wm0 + i * 16 + lm][0];
      ((v8bf*)&afrag[i])[0] = *(const v8bf*)(p + lh * 8);
      ((v8bf*)&afrag[i])[1] = *(const v8bf*)(p + 16 + lh * 8);
    }
    // B 32x16 frag: lane half h -> K {8h..8h+7} in elems 0-7, {16+8h..} in 8-15 for column lm
    v16bf bfrag[4];
#pragma unroll
    for (int j = 0; j < 4; ++j) {
      const __bf16* p = &Bs[cur][wn0 + j * 16 + lm][0];
      ((v8bf*)&bfrag[j])[0] = *(const v8bf*)(p + lh * 8);
      ((v8bf*)&bfrag[j])[1] = *(const v8bf*)(p + 16 + lh * 8);
    }

#pragma unroll
    for (int i = 0; i < 2; ++i)
#pragma unroll
      for (int j = 0; j < 4; ++j)
        acc[i][j] = __builtin_amdgcn_wmma_f32_16x16x32_bf16(
            false, afrag[i], false, bfrag[j], (short)0, acc[i][j], false, false);

    if (t + 1 < TSTEPS) ldsStore(cur ^ 1);
    __syncthreads();
  }

  // epilogue: C/D layout lane(0-15): N=lane, M=r ; lane(16-31): M=8+r.
  // Single per-lane base; i/j/r offsets are compile-time -> store immediates.
  const int row0 = tile_m + wm0 + lh * 8;
  const int col0 = tile_n + wn0 + lm;
  if constexpr (C_BF16) {
    __bf16* p = Cb + (size_t)row0 * NN + col0;
#pragma unroll
    for (int i = 0; i < 2; ++i)
#pragma unroll
      for (int j = 0; j < 4; ++j)
#pragma unroll
        for (int r = 0; r < 8; ++r) {
          float v = acc[i][j][r];
          if (ELU) v = v > 0.f ? v : (__expf(v) - 1.f);
          p[(i * 16 + r) * NN + j * 16] = (__bf16)v;
        }
  } else {
    float* p = Cf + (size_t)row0 * NN + col0;
#pragma unroll
    for (int i = 0; i < 2; ++i)
#pragma unroll
      for (int j = 0; j < 4; ++j)
#pragma unroll
        for (int r = 0; r < 8; ++r) {
          float v = acc[i][j][r];
          if (ELU) v = v > 0.f ? v : (__expf(v) - 1.f);
          p[(i * 16 + r) * NN + j * 16] = v;
        }
  }
}

// ---------------------------------------------------------------------------
// scores: per row (b,i), six length-F dot products -> rank-1 logit factors
//   s_src_SS = Wh.a1ss           s_dst_SS = Wh.a2ss + h_x.c_x + kx
//   s_src_DS = Wh.a1ds           s_dst_DS = Wh.a2ds + h_o.c_o + ko
// ---------------------------------------------------------------------------
__global__ __launch_bounds__(256) void scores_kernel(
    const __bf16* __restrict__ Wh,
    const float* __restrict__ hx, const float* __restrict__ ho,
    const float* __restrict__ aSS, const float* __restrict__ aDS,
    const float* __restrict__ cx,  const float* __restrict__ co,
    const float* __restrict__ kk,
    float* __restrict__ sSrcSS, float* __restrict__ sDstSS,
    float* __restrict__ sSrcDS, float* __restrict__ sDstDS,
    int N, int F)
{
  const int i = blockIdx.x, b = blockIdx.y, tid = threadIdx.x;
  const size_t row = ((size_t)b * N + i) * F;
  float p0 = 0, p1 = 0, p2 = 0, p3 = 0, p4 = 0, p5 = 0;
  for (int f = tid; f < F; f += 256) {
    float w = (float)Wh[row + f];
    p0 += w * aSS[f];
    p1 += w * aSS[F + f];
    p2 += w * aDS[f];
    p3 += w * aDS[F + f];
    p4 += hx[row + f] * cx[f];
    p5 += ho[row + f] * co[f];
  }
  __shared__ float red[6][256];
  red[0][tid] = p0; red[1][tid] = p1; red[2][tid] = p2;
  red[3][tid] = p3; red[4][tid] = p4; red[5][tid] = p5;
  for (int s = 128; s > 0; s >>= 1) {
    __syncthreads();
    if (tid < s)
#pragma unroll
      for (int q = 0; q < 6; ++q) red[q][tid] += red[q][tid + s];
  }
  __syncthreads();
  if (tid == 0) {
    const size_t o = (size_t)b * N + i;
    sSrcSS[o] = red[0][0];
    sDstSS[o] = red[1][0] + red[4][0] + kk[0];
    sSrcDS[o] = red[2][0];
    sDstDS[o] = red[3][0] + red[5][0] + kk[1];
  }
}

// ---------------------------------------------------------------------------
// softmax: one block per row (b,i); e = lrelu(sx_i+dx_j)*xm + lrelu(so_i+do_j)*om
//          masked by xm>0 with -9e15, row softmax, store att as bf16.
// ---------------------------------------------------------------------------
__global__ __launch_bounds__(256) void softmax_kernel(
    const float* __restrict__ xmask, const float* __restrict__ omask,
    const float* __restrict__ sSrcSS, const float* __restrict__ sDstSS,
    const float* __restrict__ sSrcDS, const float* __restrict__ sDstDS,
    __bf16* __restrict__ att, int N)
{
  const int i = blockIdx.x, b = blockIdx.y, tid = threadIdx.x;
  const size_t base = ((size_t)b * N + i) * N;
  const size_t sb   = (size_t)b * N;
  const float ssx = sSrcSS[sb + i];
  const float sso = sSrcDS[sb + i];

  float val[4];
  float vmax = -__builtin_inff();
#pragma unroll
  for (int q = 0; q < 4; ++q) {
    const int j = tid + q * 256;
    const float xm = xmask[base + j];
    const float om = omask[base + j];
    float ex = ssx + sDstSS[sb + j]; ex = ex > 0.f ? ex : GAT_ALPHA * ex;
    float eo = sso + sDstDS[sb + j]; eo = eo > 0.f ? eo : GAT_ALPHA * eo;
    const float e = ex * xm + eo * om;
    val[q] = (xm > 0.f) ? e : GAT_NEG_INF;
    vmax = fmaxf(vmax, val[q]);
  }
  __shared__ float red[256];
  red[tid] = vmax;
  for (int s = 128; s > 0; s >>= 1) {
    __syncthreads();
    if (tid < s) red[tid] = fmaxf(red[tid], red[tid + s]);
  }
  __syncthreads();
  vmax = red[0];
  __syncthreads();

  float sum = 0.f;
#pragma unroll
  for (int q = 0; q < 4; ++q) { val[q] = __expf(val[q] - vmax); sum += val[q]; }
  red[tid] = sum;
  for (int s = 128; s > 0; s >>= 1) {
    __syncthreads();
    if (tid < s) red[tid] += red[tid + s];
  }
  __syncthreads();
  const float inv = 1.f / red[0];
#pragma unroll
  for (int q = 0; q < 4; ++q)
    att[base + tid + q * 256] = (__bf16)(val[q] * inv);
}

// ---------------------------------------------------------------------------
extern "C" void kernel_launch(void* const* d_in, const int* in_sizes, int n_in,
                              void* d_out, int out_size, void* d_ws, size_t ws_size,
                              hipStream_t stream) {
  (void)in_sizes; (void)n_in; (void)out_size; (void)ws_size;
  const int BZ = 16, N = 1024, F = 1024;

  const float* h     = (const float*)d_in[0];
  const float* h_x   = (const float*)d_in[1];
  const float* h_o   = (const float*)d_in[2];
  const float* xmask = (const float*)d_in[3];
  const float* omask = (const float*)d_in[4];
  const float* W     = (const float*)d_in[5];
  const float* aSS   = (const float*)d_in[6];
  const float* aDS   = (const float*)d_in[7];
  const float* cskW  = (const float*)d_in[8];
  const float* cskB  = (const float*)d_in[9];
  float* out = (float*)d_out;

  // workspace layout
  const size_t elems = (size_t)BZ * N * F;          // 16M
  char* ws = (char*)d_ws;
  __bf16* Whb  = (__bf16*)ws;                       // 32 MB bf16
  __bf16* att  = (__bf16*)(ws + elems * 2);         // 32 MB bf16
  float*  sbuf = (float*)(ws + elems * 4);
  float* s0 = sbuf;                 // s_src_SS  [BZ*N]
  float* s1 = s0 + (size_t)BZ * N;  // s_dst_SS
  float* s2 = s1 + (size_t)BZ * N;  // s_src_DS
  float* s3 = s2 + (size_t)BZ * N;  // s_dst_DS
  float* cx = s3 + (size_t)BZ * N;  // [F]
  float* co = cx + F;               // [F]
  float* kk = co + F;               // [2]

  // 1) fold csk_W / csk_b into per-feature vectors (kills two 34-GFLOP GEMMs)
  prep_kernel<<<dim3((F + 255) / 256), 256, 0, stream>>>(cskW, cskB, aSS, aDS, cx, co, kk, F);

  // 2) Whb = bf16(h @ W)  (batched over z; B shared -> strideB = 0)
  gemm_wmma<false, false, true, false><<<dim3(N / 128, N / 128, BZ), 256, 0, stream>>>(
      (const void*)h, (const void*)W, (void*)Whb,
      (long long)N * F, 0, (long long)N * F);

  // 3) rank-1 logit factors
  scores_kernel<<<dim3(N, BZ), 256, 0, stream>>>(
      Whb, h_x, h_o, aSS, aDS, cx, co, kk, s0, s1, s2, s3, N, F);

  // 4) masked row softmax -> att (bf16)
  softmax_kernel<<<dim3(N, BZ), 256, 0, stream>>>(
      xmask, omask, s0, s1, s2, s3, att, N);

  // 5) out = elu(att @ Whb)  (bf16 A and B, elu epilogue, f32 out)
  gemm_wmma<true, true, false, true><<<dim3(N / 128, N / 128, BZ), 256, 0, stream>>>(
      (const void*)att, (const void*)Whb, (void*)out,
      (long long)N * N, (long long)N * F, (long long)N * F);
}
